// TransitionForest_71416716198537
// MI455X (gfx1250) — matrix operand order repaired and checked by reading, not compiled, not run on hardware
//
#include <hip/hip_runtime.h>
#include <math.h>

// ---------------- problem sizes ----------------
#define B_   64
#define S_   256
#define A_   256
#define H_   512
#define WD_  320      // TE+PE
#define G4_  2048     // 4*H
#define NA_  75
#define NAP_ 80       // NA padded to /16
#define F5H_ 2560     // 5*H
#define BH_  (B_*H_)  // 32768

typedef __attribute__((ext_vector_type(16))) _Float16 v16h;
typedef __attribute__((ext_vector_type(8)))  _Float16 v8h;
typedef __attribute__((ext_vector_type(8)))  float    v8f;

__device__ __forceinline__ float sigf(float x){ return 1.0f/(1.0f + expf(-x)); }

// A fragment: 16x32 (MxK) f16, row-major A, lda=K. ISA layout:
// lane L: m=L%16 ; elems 0..7 -> k = (L/16)*8 + i ; elems 8..15 -> k = 16 + (L/16)*8 + i
__device__ __forceinline__ v16h load_a_frag(const _Float16* __restrict__ A, int lda,
                                            int m0, int k0, int lane){
  const _Float16* p = A + (size_t)(m0 + (lane & 15)) * lda + (k0 + ((lane >> 4) << 3));
  v8h lo = *(const v8h*)(p);
  v8h hi = *(const v8h*)(p + 16);
  v16h r;
#pragma unroll
  for (int i = 0; i < 8; ++i) { r[i] = lo[i]; r[i+8] = hi[i]; }
  return r;
}

// B fragment: 32x16 (KxN); source W is row-major (N,K) (we compute x @ W^T).
// lane L: n=L%16 ; elems 0..7 -> k = (L/16)*8 + i ; elems 8..15 -> k = 16 + (L/16)*8 + i
__device__ __forceinline__ v16h load_b_frag(const _Float16* __restrict__ W, int ldw,
                                            int n0, int k0, int lane){
  const _Float16* p = W + (size_t)(n0 + (lane & 15)) * ldw + (k0 + ((lane >> 4) << 3));
  v8h lo = *(const v8h*)(p);
  v8h hi = *(const v8h*)(p + 16);
  v16h r;
#pragma unroll
  for (int i = 0; i < 8; ++i) { r[i] = lo[i]; r[i+8] = hi[i]; }
  return r;
}

// ---------------- small utility kernels ----------------
__global__ __launch_bounds__(256) void k_f32_to_f16(const float* __restrict__ s,
                                                    _Float16* __restrict__ d, long n){
  long i = (long)blockIdx.x*256 + threadIdx.x;
  if (i < n) d[i] = (_Float16)s[i];
}
__global__ __launch_bounds__(256) void k_zero_f16(_Float16* __restrict__ p, long n){
  long i = (long)blockIdx.x*256 + threadIdx.x;
  if (i < n) p[i] = (_Float16)0.0f;
}
__global__ __launch_bounds__(256) void k_zero_f32(float* __restrict__ p, long n){
  long i = (long)blockIdx.x*256 + threadIdx.x;
  if (i < n) p[i] = 0.0f;
}
// f2a_W (75x512) -> padded f16 (80x512), pad rows zero
__global__ __launch_bounds__(256) void k_pad_f2a(const float* __restrict__ W,
                                                 _Float16* __restrict__ Wp){
  int i = blockIdx.x*256 + threadIdx.x;
  if (i >= NAP_*H_) return;
  int r = i / H_, c = i - (i/H_)*H_;
  Wp[i] = (r < NA_) ? (_Float16)W[r*H_ + c] : (_Float16)0.0f;
}
// tok f16 laid out (S,B,WD): row = s*64+b
__global__ __launch_bounds__(256) void k_embed_tok(const int* __restrict__ tokb,
    const int* __restrict__ posb, const float* __restrict__ wemb,
    const float* __restrict__ pemb, _Float16* __restrict__ out){
  long idx = (long)blockIdx.x*256 + threadIdx.x;
  if (idx >= (long)S_*B_*WD_) return;
  int k = (int)(idx % WD_);
  long sb = idx / WD_;
  int b = (int)(sb % B_), s = (int)(sb / B_);
  float v;
  if (k < 256) v = wemb[(size_t)tokb[b*S_+s]*256 + k];
  else         v = pemb[(size_t)posb[b*S_+s]*64 + (k-256)];
  out[idx] = (_Float16)v;
}
// a_emb f16 laid out (A,B,64): row = t*64+b
__global__ __launch_bounds__(256) void k_gather_aemb(const int* __restrict__ ab,
    const float* __restrict__ aemb, _Float16* __restrict__ out){
  long idx = (long)blockIdx.x*256 + threadIdx.x;
  if (idx >= (long)A_*B_*64) return;
  int e = (int)(idx % 64);
  long tb = idx / 64;
  int b = (int)(tb % B_), t = (int)(tb / B_);
  out[idx] = (_Float16)aemb[(size_t)ab[b*A_+t]*64 + e];
}
// composed[t*64+b][i] = tanh( red_W[lab] (320x640) . pair + red_b[lab] ), f16 out
__global__ __launch_bounds__(256) void k_composed(const _Float16* __restrict__ tokh,
    const int* __restrict__ ab, const float* __restrict__ redW,
    const float* __restrict__ redb, _Float16* __restrict__ out){
  __shared__ float pv[2*WD_];
  int tb = blockIdx.x;
  int t = tb / B_, b = tb - (tb/B_)*B_;
  int lab = ab[b*A_ + t] % 48;
  for (int j = threadIdx.x; j < 2*WD_; j += 256){
    int tt = (j < WD_) ? t : (t > 0 ? t-1 : 0);
    int jj = (j < WD_) ? j : j - WD_;
    pv[j] = (float)tokh[((size_t)tt*B_ + b)*WD_ + jj];
  }
  __syncthreads();
  const float* Wl = redW + (size_t)lab*WD_*2*WD_;
  for (int i = threadIdx.x; i < WD_; i += 256){
    float sacc = redb[lab*WD_ + i];
    const float* wr = Wl + (size_t)i*2*WD_;
    for (int j = 0; j < 2*WD_; ++j) sacc += wr[j]*pv[j];
    out[((size_t)t*B_ + b)*WD_ + i] = (_Float16)tanhf(sacc);
  }
}

// ---------------- generic WMMA GEMM:  C(MxN) = A(MxK) @ W(NxK)^T (+bias)(+addsrc) ----------------
__global__ __launch_bounds__(256) void k_gemm(const _Float16* __restrict__ A,
    const _Float16* __restrict__ Bw, const float* __restrict__ bias,
    const float* __restrict__ addsrc, float* __restrict__ Cf,
    _Float16* __restrict__ Ch, int M, int N, int K, int act)
{
  int gw   = (int)((blockIdx.x * 256 + threadIdx.x) >> 5);
  int lane = threadIdx.x & 31;
  int ntiles = N >> 4;
  int total  = (M >> 4) * ntiles;
  if (gw >= total) return;                       // uniform per wave
  int mt = gw / ntiles, nt = gw - mt * ntiles;
  int m0 = mt << 4, n0 = nt << 4;
  float bv = bias ? bias[n0 + (lane & 15)] : 0.0f;
  v8f acc;
#pragma unroll
  for (int v = 0; v < 8; ++v) acc[v] = bv;
  for (int k0 = 0; k0 < K; k0 += 32) {
    v16h af = load_a_frag(A,  K, m0, k0, lane);
    v16h bf = load_b_frag(Bw, K, n0, k0, lane);
    acc = __builtin_amdgcn_wmma_f32_16x16x32_f16(false, af, false, bf,
                                                 (short)0, acc, false, false);
  }
#pragma unroll
  for (int v = 0; v < 8; ++v) {
    int m = v + ((lane >> 4) << 3);
    int n = lane & 15;
    size_t idx = (size_t)(m0 + m) * N + (n0 + n);
    float x = acc[v];
    if (addsrc) x += addsrc[idx];
    if (act == 1) x = fmaxf(x, 0.0f);
    else if (act == 2) x = tanhf(x);
    if (Cf) Cf[idx] = x;
    if (Ch) Ch[idx] = (_Float16)x;
  }
}

// ---------------- fused recurrent LSTM step body ----------------
// gates(64x2048) = G[grow0..] + hin(64x512) @ Whh(2048x512)^T ; then pointwise update.
// blk: 0..63 = 4 m-tiles x 16 j-pairs ; 8 waves = 4 gates x 2 j-tiles.
__device__ __forceinline__ void lstm_tile_body(float* gsh,
    const _Float16* __restrict__ G, int grow0,
    const _Float16* __restrict__ Whh,
    const _Float16* __restrict__ hin,
    float* __restrict__ c, _Float16* __restrict__ hout,
    float* __restrict__ out_f32, _Float16* __restrict__ out_f16, int orow0, int blk)
{
  int w = threadIdx.x >> 5, lane = threadIdx.x & 31;
  int mt = blk >> 4, jp = blk & 15;
  int gsel = w & 3, jhalf = w >> 2;
  int m0 = mt << 4;
  int n0 = gsel*H_ + ((jp*2 + jhalf) << 4);
  v8f acc;
#pragma unroll
  for (int v = 0; v < 8; ++v){
    int m = v + ((lane >> 4) << 3);
    int n = lane & 15;
    acc[v] = (float)G[(size_t)(grow0 + m0 + m)*G4_ + n0 + n];
  }
  for (int k0 = 0; k0 < H_; k0 += 32){
    if (k0 + 128 < H_)   // pull weight lines into WGP$ ahead of the load clauses
      __builtin_prefetch((const void*)(Whh + (size_t)(n0 + (lane & 15))*H_ + k0 + 128), 0, 0);
    v16h af = load_a_frag(hin, H_, m0, k0, lane);
    v16h bf = load_b_frag(Whh, H_, n0, k0, lane);
    acc = __builtin_amdgcn_wmma_f32_16x16x32_f16(false, af, false, bf,
                                                 (short)0, acc, false, false);
  }
#pragma unroll
  for (int v = 0; v < 8; ++v){
    int m = v + ((lane >> 4) << 3);
    int n = lane & 15;
    gsh[(gsel*16 + m)*32 + (jhalf*16 + n)] = acc[v];
  }
  __syncthreads();
  for (int idx = threadIdx.x; idx < 512; idx += 256){
    int m = idx >> 5, jl = idx & 31;
    int b = m0 + m;
    int j = jp*32 + jl;
    float gi = gsh[(0*16+m)*32+jl];
    float gf = gsh[(1*16+m)*32+jl];
    float gg = gsh[(2*16+m)*32+jl];
    float go = gsh[(3*16+m)*32+jl];
    float cv = c[b*H_+j];
    cv = sigf(gf)*cv + sigf(gi)*tanhf(gg);
    float hv = sigf(go)*tanhf(cv);
    c[b*H_+j] = cv;
    hout[b*H_+j] = (_Float16)hv;
    if (out_f32) out_f32[(size_t)(orow0+b)*H_ + j] = hv;
    if (out_f16) out_f16[(size_t)(orow0+b)*H_ + j] = (_Float16)hv;
  }
}

// token LSTM: one cell, also emits tok_out (f32 + f16)
__global__ __launch_bounds__(256) void k_lstm_step(
    const _Float16* __restrict__ G, int grow0,
    const _Float16* __restrict__ Whh, const _Float16* __restrict__ hin,
    float* __restrict__ c, _Float16* __restrict__ hout,
    float* __restrict__ out_f32, _Float16* __restrict__ out_f16, int orow0)
{
  __shared__ float gsh[4*16*32];
  lstm_tile_body(gsh, G, grow0, Whh, hin, c, hout, out_f32, out_f16, orow0, blockIdx.x);
}

// action step: the two independent LSTM cells fused into one 128-block launch
__global__ __launch_bounds__(256) void k_lstm_step2(
    const _Float16* __restrict__ Gac, const _Float16* __restrict__ Gst, int grow0,
    const _Float16* __restrict__ acWhh, const _Float16* __restrict__ stWhh,
    const _Float16* __restrict__ ha_in, float* __restrict__ ca, _Float16* __restrict__ ha_out,
    const _Float16* __restrict__ hs_in, float* __restrict__ cs, _Float16* __restrict__ hs_out)
{
  __shared__ float gsh[4*16*32];
  int cell = blockIdx.x >> 6;          // 0 = action cell, 1 = state cell
  int blk  = blockIdx.x & 63;
  const _Float16* G   = cell ? Gst   : Gac;
  const _Float16* Whh = cell ? stWhh : acWhh;
  const _Float16* hin = cell ? hs_in : ha_in;
  float* c            = cell ? cs    : ca;
  _Float16* hout      = cell ? hs_out: ha_out;
  lstm_tile_body(gsh, G, grow0, Whh, hin, c, hout, nullptr, nullptr, 0, blk);
}

// hid = relu(feat @ h2f_W^T + b); feat = [hs_new|hs_prev|h1|buf_t|ha] gathered inline
// (each 512-wide K-segment is a whole number of K=32 steps -> source select is uniform)
__global__ __launch_bounds__(256) void k_hid(
    const _Float16* __restrict__ hsn, const _Float16* __restrict__ hsp,
    const _Float16* __restrict__ h1,  const _Float16* __restrict__ buf,
    const _Float16* __restrict__ ha,
    const _Float16* __restrict__ h2fW, const float* __restrict__ h2fb,
    _Float16* __restrict__ hid)
{
  int gw = (blockIdx.x*256 + threadIdx.x) >> 5;   // 0..127 = 4 m-tiles x 32 n-tiles
  int lane = threadIdx.x & 31;
  int mt = gw >> 5, nt = gw & 31;
  int m0 = mt << 4, n0 = nt << 4;
  float bv = h2fb[n0 + (lane & 15)];
  v8f acc;
#pragma unroll
  for (int v = 0; v < 8; ++v) acc[v] = bv;
#pragma unroll
  for (int seg = 0; seg < 5; ++seg){
    const _Float16* s;
    switch (seg){ case 0: s = hsn; break; case 1: s = hsp; break;
                  case 2: s = h1;  break; case 3: s = buf; break; default: s = ha; }
    for (int k0 = 0; k0 < H_; k0 += 32){
      v16h af = load_a_frag(s, H_, m0, k0, lane);
      v16h bf = load_b_frag(h2fW, F5H_, n0, seg*H_ + k0, lane);
      acc = __builtin_amdgcn_wmma_f32_16x16x32_f16(false, af, false, bf,
                                                   (short)0, acc, false, false);
    }
  }
#pragma unroll
  for (int v = 0; v < 8; ++v){
    int m = v + ((lane >> 4) << 3);
    int n = lane & 15;
    float x = fmaxf(acc[v], 0.0f);
    hid[(size_t)(m0 + m)*H_ + n0 + n] = (_Float16)x;
  }
}

// logits (64x80 via WMMA, staged in LDS) + log-softmax NLL, single workgroup
__global__ __launch_bounds__(256) void k_head(
    const _Float16* __restrict__ hid, const _Float16* __restrict__ f2aW,
    const float* __restrict__ f2ab, const int* __restrict__ ab, int t,
    float* __restrict__ losses)
{
  __shared__ float lg[B_*NAP_];     // 64x80 logits (no bias)
  int w = threadIdx.x >> 5, lane = threadIdx.x & 31;
  for (int tile = w; tile < 20; tile += 8){      // 4 m-tiles x 5 n-tiles
    int mt = tile / 5, nt = tile - (tile/5)*5;
    int m0 = mt << 4, n0 = nt << 4;
    v8f acc;
#pragma unroll
    for (int v = 0; v < 8; ++v) acc[v] = 0.0f;
    for (int k0 = 0; k0 < H_; k0 += 32){
      v16h af = load_a_frag(hid,  H_, m0, k0, lane);
      v16h bf = load_b_frag(f2aW, H_, n0, k0, lane);
      acc = __builtin_amdgcn_wmma_f32_16x16x32_f16(false, af, false, bf,
                                                   (short)0, acc, false, false);
    }
#pragma unroll
    for (int v = 0; v < 8; ++v){
      int m = v + ((lane >> 4) << 3);
      int n = lane & 15;
      lg[(m0 + m)*NAP_ + n0 + n] = acc[v];
    }
  }
  __syncthreads();
  int b = threadIdx.x;
  if (b < B_){
    const float* lr = lg + b*NAP_;
    float mx = -1e30f;
    for (int n = 0; n < NA_; ++n) mx = fmaxf(mx, lr[n] + f2ab[n]);
    float se = 0.0f;
    for (int n = 0; n < NA_; ++n) se += expf(lr[n] + f2ab[n] - mx);
    int act = ab[b*A_ + t];
    losses[t*B_ + b] = (mx + logf(se)) - (lr[act] + f2ab[act]);
  }
}

__global__ __launch_bounds__(256) void k_mean(const float* __restrict__ losses,
                                              float* __restrict__ out){
  __shared__ float sm[256];
  float acc = 0.0f;
  for (int i = threadIdx.x; i < A_*B_; i += 256) acc += losses[i];
  sm[threadIdx.x] = acc; __syncthreads();
  for (int o = 128; o > 0; o >>= 1){
    if ((int)threadIdx.x < o) sm[threadIdx.x] += sm[threadIdx.x + o];
    __syncthreads();
  }
  if (threadIdx.x == 0) out[0] = sm[0] / (float)(A_*B_);
}

// ---------------- host ----------------
static inline int gemm_grid(int M, int N){ return (((M >> 4)*(N >> 4)) + 7) / 8; }

extern "C" void kernel_launch(void* const* d_in, const int* in_sizes, int n_in,
                              void* d_out, int out_size, void* d_ws, size_t ws_size,
                              hipStream_t stream)
{
  (void)in_sizes; (void)n_in; (void)out_size; (void)ws_size;
  const int*   token_batch  = (const int*)  d_in[0];
  const int*   pos_batch    = (const int*)  d_in[1];
  const int*   action_batch = (const int*)  d_in[3];
  const float* word_emb = (const float*)d_in[4];
  const float* pos_emb  = (const float*)d_in[5];
  const float* act_emb  = (const float*)d_in[6];
  const float* tok_Wih = (const float*)d_in[7];
  const float* tok_Whh = (const float*)d_in[8];
  const float* tok_b   = (const float*)d_in[9];
  const float* st_Wih  = (const float*)d_in[10];
  const float* st_Whh  = (const float*)d_in[11];
  const float* st_b    = (const float*)d_in[12];
  const float* ac_Wih  = (const float*)d_in[13];
  const float* ac_Whh  = (const float*)d_in[14];
  const float* ac_b    = (const float*)d_in[15];
  const float* red_W   = (const float*)d_in[16];
  const float* red_b   = (const float*)d_in[17];
  const float* comp_W  = (const float*)d_in[18];
  const float* comp_b  = (const float*)d_in[19];
  const float* h2f_W   = (const float*)d_in[20];
  const float* h2f_b   = (const float*)d_in[21];
  const float* f2a_W   = (const float*)d_in[22];
  const float* f2a_b   = (const float*)d_in[23];

  char* base = (char*)d_ws; size_t off = 0;
  auto alloc = [&](size_t bytes)->char*{
    char* p = base + off; off = (off + bytes + 255) & ~(size_t)255; return p;
  };
  const size_t MB = (size_t)S_*B_;   // 16384 rows, == A_*B_
  _Float16* tokWih_h = (_Float16*)alloc((size_t)G4_*WD_*2);
  _Float16* tokWhh_h = (_Float16*)alloc((size_t)G4_*H_*2);
  _Float16* stWih_h  = (_Float16*)alloc((size_t)G4_*H_*2);
  _Float16* stWhh_h  = (_Float16*)alloc((size_t)G4_*H_*2);
  _Float16* acWih_h  = (_Float16*)alloc((size_t)G4_*64*2);
  _Float16* acWhh_h  = (_Float16*)alloc((size_t)G4_*H_*2);
  _Float16* compW_h  = (_Float16*)alloc((size_t)H_*WD_*2);
  _Float16* h2fW_h   = (_Float16*)alloc((size_t)H_*F5H_*2);
  _Float16* f2aW_h   = (_Float16*)alloc((size_t)NAP_*H_*2);
  _Float16* tok_h    = (_Float16*)alloc(MB*WD_*2);
  _Float16* aemb_h   = (_Float16*)alloc(MB*64*2);
  _Float16* Gtok_h   = (_Float16*)alloc(MB*G4_*2);
  _Float16* Gac_h    = (_Float16*)alloc(MB*G4_*2);
  _Float16* Gst_h    = (_Float16*)alloc(MB*G4_*2);
  _Float16* comp_h   = (_Float16*)alloc(MB*WD_*2);
  float*    tok_out  = (float*)   alloc(MB*H_*4);
  _Float16* tok_outh = (_Float16*)alloc(MB*H_*2);
  _Float16* stin_h   = (_Float16*)alloc(MB*H_*2);
  _Float16* htok0    = (_Float16*)alloc((size_t)BH_*2);
  _Float16* htok1    = (_Float16*)alloc((size_t)BH_*2);
  float*    ctok     = (float*)   alloc((size_t)BH_*4);
  _Float16* ha0      = (_Float16*)alloc((size_t)BH_*2);
  _Float16* ha1      = (_Float16*)alloc((size_t)BH_*2);
  float*    ca       = (float*)   alloc((size_t)BH_*4);
  float*    cs       = (float*)   alloc((size_t)BH_*4);
  _Float16* hist     = (_Float16*)alloc((size_t)(A_+2)*BH_*2);
  _Float16* hid_h    = (_Float16*)alloc((size_t)B_*H_*2);
  float*    losses   = (float*)   alloc(MB*4);
  _Float16* htok[2] = { htok0, htok1 };
  _Float16* habuf[2]= { ha0, ha1 };

  auto conv = [&](const float* s, _Float16* d, size_t n){
    k_f32_to_f16<<<(int)((n + 255)/256), 256, 0, stream>>>(s, d, (long)n);
  };
  auto z16 = [&](_Float16* p, size_t n){
    k_zero_f16<<<(int)((n + 255)/256), 256, 0, stream>>>(p, (long)n);
  };
  auto z32 = [&](float* p, size_t n){
    k_zero_f32<<<(int)((n + 255)/256), 256, 0, stream>>>(p, (long)n);
  };

  // --- weight conversion (weights live in L2 thereafter) ---
  conv(tok_Wih, tokWih_h, (size_t)G4_*WD_);
  conv(tok_Whh, tokWhh_h, (size_t)G4_*H_);
  conv(st_Wih,  stWih_h,  (size_t)G4_*H_);
  conv(st_Whh,  stWhh_h,  (size_t)G4_*H_);
  conv(ac_Wih,  acWih_h,  (size_t)G4_*64);
  conv(ac_Whh,  acWhh_h,  (size_t)G4_*H_);
  conv(comp_W,  compW_h,  (size_t)H_*WD_);
  conv(h2f_W,   h2fW_h,   (size_t)H_*F5H_);
  k_pad_f2a<<<(NAP_*H_ + 255)/256, 256, 0, stream>>>(f2a_W, f2aW_h);

  // --- embeddings / gathers ---
  k_embed_tok<<<(int)((MB*WD_ + 255)/256), 256, 0, stream>>>(token_batch, pos_batch,
                                                             word_emb, pos_emb, tok_h);
  k_gather_aemb<<<(int)((MB*64 + 255)/256), 256, 0, stream>>>(action_batch, act_emb, aemb_h);

  // --- all non-recurrent matmuls as big parallel WMMA GEMMs ---
  k_gemm<<<gemm_grid((int)MB, G4_), 256, 0, stream>>>(tok_h, tokWih_h, tok_b, nullptr,
        nullptr, Gtok_h, (int)MB, G4_, WD_, 0);                       // G_tok
  k_gemm<<<gemm_grid((int)MB, G4_), 256, 0, stream>>>(aemb_h, acWih_h, ac_b, nullptr,
        nullptr, Gac_h, (int)MB, G4_, 64, 0);                         // G_ac
  k_composed<<<(int)MB, 256, 0, stream>>>(tok_h, action_batch, red_W, red_b, comp_h);

  // --- token LSTM serial chain (one fused WMMA+pointwise kernel per step) ---
  z16(htok[0], (size_t)BH_);
  z32(ctok, (size_t)BH_);
  for (int s = 0; s < S_; ++s){
    k_lstm_step<<<64, 256, 0, stream>>>(Gtok_h, s*B_, tokWhh_h,
        htok[s & 1], ctok, htok[(s + 1) & 1], tok_out, tok_outh, s*B_);
  }

  // --- st_in = tok_out + composed @ comp_W^T + comp_b ; then G_st ---
  k_gemm<<<gemm_grid((int)MB, H_), 256, 0, stream>>>(comp_h, compW_h, comp_b, tok_out,
        nullptr, stin_h, (int)MB, H_, WD_, 0);
  k_gemm<<<gemm_grid((int)MB, G4_), 256, 0, stream>>>(stin_h, stWih_h, st_b, nullptr,
        nullptr, Gst_h, (int)MB, G4_, H_, 0);

  // --- action scan: 3 launches per step ---
  z16(habuf[0], (size_t)BH_);
  z32(ca, (size_t)BH_);
  z32(cs, (size_t)BH_);
  z16(hist, (size_t)2*BH_);       // hs at t=-2,-1
  for (int t = 0; t < A_; ++t){
    _Float16* hs_m2 = hist + (size_t)t      * BH_;
    _Float16* hs_m1 = hist + (size_t)(t + 1)* BH_;
    _Float16* hs_0  = hist + (size_t)(t + 2)* BH_;
    k_lstm_step2<<<128, 256, 0, stream>>>(Gac_h, Gst_h, t*B_, acWhh_h, stWhh_h,
        habuf[t & 1], ca, habuf[(t + 1) & 1], hs_m1, cs, hs_0);
    k_hid<<<16, 256, 0, stream>>>(hs_0, hs_m1, hs_m2, tok_outh + (size_t)t*BH_,
        habuf[(t + 1) & 1], h2fW_h, h2f_b, hid_h);
    k_head<<<1, 256, 0, stream>>>(hid_h, f2aW_h, f2a_b, action_batch, t, losses);
  }

  k_mean<<<1, 256, 0, stream>>>(losses, (float*)d_out);
}
